// TokenMixWeightGenerator_43748536877627
// MI455X (gfx1250) — matrix operand order repaired and checked by reading, not compiled, not run on hardware
//
#include <hip/hip_runtime.h>
#include <math.h>

// ---------------------------------------------------------------------------
// Problem constants (match reference): B=2, L=1024, D=512, H=8, Dh=64, Q=64
// ---------------------------------------------------------------------------
#define BB   2
#define LL   1024
#define DD   512
#define HH   8
#define DH   64
#define QQ   64
#define NC   16   // L / Q

typedef __attribute__((ext_vector_type(16))) __bf16 v16bf;
typedef __attribute__((ext_vector_type(8)))  __bf16 v8bf;
typedef __attribute__((ext_vector_type(8)))  float  v8f;

__device__ __forceinline__ __bf16 f2bf(float f) { return (__bf16)f; }

// fast sigmoid: one v_exp_f32 + one v_rcp_f32
__device__ __forceinline__ float fast_sigmoid(float v) {
    return __builtin_amdgcn_rcpf(1.f + __expf(-v));
}

// A-matrix (16x32 bf16) lane/element -> K mapping (ISA 7.12.2):
// lanes 0-15 : M=lane,    VGPR0..3 K=0..7,   VGPR4..7 K=16..23
// lanes 16-31: M=lane-16, VGPR0..3 K=8..15,  VGPR4..7 K=24..31
// => elements 0..7 are contiguous d = base+half*8+e, elements 8..15 are
//    contiguous d = base+16+half*8+(e-8): two 16-byte chunks per lane.
__device__ __forceinline__ int a_kmap(int e, int half) {
    return (e < 8) ? (half * 8 + e) : (8 + half * 8 + e);
}
// B-matrix (32x16 bf16): same K mapping as A (lanes 0-15 K=0..7,16..23; lanes 16-31 K=8..15,24..31).
__device__ __forceinline__ int b_kmap(int e, int half) {
    return (e < 8) ? (half * 8 + e) : (8 + half * 8 + e);
}

// build v16bf from two aligned 8-element LDS vectors (two ds_load_b128)
__device__ __forceinline__ v16bf ld_a_op(const __bf16* rowp, int ks, int half) {
    const v8bf lo = *(const v8bf*)(rowp + ks * 32 + half * 8);
    const v8bf hi = *(const v8bf*)(rowp + ks * 32 + 16 + half * 8);
    v16bf r;
    for (int e = 0; e < 8; ++e) { r[e] = lo[e]; r[e + 8] = hi[e]; }
    return r;
}
__device__ __forceinline__ v16bf ld_b_op(const __bf16* rowp, int ks, int half) {
    const v8bf lo = *(const v8bf*)(rowp + ks * 32 + half * 8);
    const v8bf hi = *(const v8bf*)(rowp + ks * 32 + 16 + half * 8);
    v16bf r;
    for (int e = 0; e < 8; ++e) { r[e] = lo[e]; r[e + 8] = hi[e]; }
    return r;
}

// ---------------------------------------------------------------------------
// Kernel 1: projections  out = act(x @ W^T + b), stored in (B,H,L,Dh) layout.
// blockIdx.y selects projection (0=A sigmoid, 1=B, 2=C). 256 thr = 8 waves,
// each block computes a 64x64 tile of the (2048 x 512) output via WMMA bf16.
// ---------------------------------------------------------------------------
__global__ void proj_kernel(const float* __restrict__ x,
                            const float* __restrict__ W_A, const float* __restrict__ b_A,
                            const float* __restrict__ W_B, const float* __restrict__ b_B,
                            const float* __restrict__ W_C, const float* __restrict__ b_C,
                            float* __restrict__ Ao, float* __restrict__ Bo,
                            float* __restrict__ Co) {
    const int p = blockIdx.y;
    const float* W    = (p == 0) ? W_A : ((p == 1) ? W_B : W_C);
    const float* bias = (p == 0) ? b_A : ((p == 1) ? b_B : b_C);
    float* outp       = (p == 0) ? Ao  : ((p == 1) ? Bo  : Co);

    const int tr = blockIdx.x >> 3;  // 0..31 row tile  (64 token rows)
    const int tc = blockIdx.x & 7;   // 0..7  col tile  (64 feature cols)
    const int tid  = threadIdx.x;
    const int lane = tid & 31;
    const int wave = tid >> 5;
    const int half = lane >> 4;
    const int lrow = lane & 15;

    for (int t = wave; t < 16; t += 8) {
        const int qi = t >> 2, ki = t & 3;
        const int row = tr * 64 + qi * 16 + lrow;  // token index (A operand row)
        const int jc  = tc * 64 + ki * 16 + lrow;  // output col  (B operand col)
        v8f acc;
        for (int i = 0; i < 8; ++i) acc[i] = 0.f;
        for (int ks = 0; ks < DD / 32; ++ks) {
            v16bf av, bv;
            for (int e = 0; e < 16; ++e) {
                const int da = ks * 32 + a_kmap(e, half);
                av[e] = f2bf(x[(size_t)row * DD + da]);
                const int db = ks * 32 + b_kmap(e, half);
                bv[e] = f2bf(W[(size_t)jc * DD + db]);  // B[k,j] = W[j,k]
            }
            acc = __builtin_amdgcn_wmma_f32_16x16x32_bf16(
                false, av, false, bv, (short)0, acc, false, false);
        }
        for (int r = 0; r < 8; ++r) {
            const int i = tr * 64 + qi * 16 + r + half * 8;  // C/D layout row
            const int j = tc * 64 + ki * 16 + lrow;
            float v = acc[r] + bias[j];
            if (p == 0) v = fast_sigmoid(v);                 // sigmoid for A
            const int bi = i >> 10, l = i & (LL - 1);
            const int hh = j >> 6,  dh = j & (DH - 1);
            outp[(((size_t)bi * HH + hh) * LL + l) * DH + dh] = v;
        }
    }
}

// ---------------------------------------------------------------------------
// Kernel 2: S = cumsum_l(log A) per (b,h,d) channel. 16 blocks x 64 threads.
// ---------------------------------------------------------------------------
__global__ void scan_kernel(const float* __restrict__ A, float* __restrict__ S) {
    const int bh = blockIdx.x;      // 0..15
    const int d  = threadIdx.x;     // 0..63
    const size_t base = (size_t)bh * LL * DH;
    float acc = 0.f;
    for (int l = 0; l < LL; ++l) {
        acc += __logf(A[base + (size_t)l * DH + d]);
        S[base + (size_t)l * DH + d] = acc;
    }
}

// ---------------------------------------------------------------------------
// Kernel 3: chunked semiseparable expansion. One block per (m, h, b).
// 256 threads = 8 waves. Per source chunk n<m the rescaled bf16 operands
// (qf' = qf*Dcross, kf) are materialized ONCE in LDS in WMMA-friendly
// row-major order, so each wave's operand fetch is two ds_load_b128 per
// matrix. Diagonal chunk: strictly-lower 16x16 sub-tiles via WMMA with
// exponent-safe rescaling, diagonal sub-tiles masked element-wise, upper
// region zero-filled with float4 stores.
// ---------------------------------------------------------------------------
__global__ void mix_kernel(const float* __restrict__ C, const float* __restrict__ Bf,
                           const float* __restrict__ S, float* __restrict__ T) {
    const int m = blockIdx.x, h = blockIdx.y, b = blockIdx.z;
    const int bh = b * HH + h;
    const size_t base  = (size_t)bh * LL * DH;   // into (L,Dh) arrays
    const size_t obase = (size_t)bh * LL * LL;   // into (L,L) output
    const int tid  = threadIdx.x;
    const int lane = tid & 31;
    const int wave = tid >> 5;
    const int half = lane >> 4;
    const int lrow = lane & 15;

    __shared__ float  s_qf [QQ][DH];    // C * exp(Sm1 - Eprev_m)      (f32)
    __shared__ float  s_s  [QQ][DH];    // S chunk n                   (f32)
    __shared__ __bf16 s_kf [QQ][DH];    // B * exp(E_n - S)            (bf16)
    __shared__ __bf16 s_qfb[QQ][DH];    // qf * exp(Eprev_m - E_n)     (bf16)
    __shared__ float  s_dc[DH];         // exp(Eprev_m - E_n)
    __shared__ float  s_ep[DH];         // Eprev_m

    if (tid < DH) s_ep[tid] = (m == 0) ? 0.f : S[base + (size_t)(m * QQ - 1) * DH + tid];
    __syncthreads();
    for (int idx = tid; idx < QQ * DH; idx += 256) {
        const int q = idx >> 6, d = idx & 63;
        const int l = m * QQ + q;
        const float sm1 = (l == 0) ? 0.f : S[base + (size_t)(l - 1) * DH + d];
        s_qf[q][d] = C[base + (size_t)l * DH + d] * __expf(sm1 - s_ep[d]);
    }

    for (int n = 0; n <= m; ++n) {
        __syncthreads();  // previous iteration fully consumed LDS buffers
        for (int idx = tid; idx < QQ * DH; idx += 256) {
            const int k = idx >> 6, d = idx & 63;
            s_s[k][d] = S[base + (size_t)(n * QQ + k) * DH + d];
        }
        __syncthreads();

        if (n < m) {
            // ---- stage bf16 operands once per chunk pair ----
            if (tid < DH) s_dc[tid] = __expf(s_ep[tid] - s_s[QQ - 1][tid]);
            __syncthreads();
            for (int idx = tid; idx < QQ * DH; idx += 256) {
                const int k = idx >> 6, d = idx & 63;
                const float bg = Bf[base + (size_t)(n * QQ + k) * DH + d];
                s_kf [k][d] = f2bf(bg * __expf(s_s[QQ - 1][d] - s_s[k][d]));
                s_qfb[k][d] = f2bf(s_qf[k][d] * s_dc[d]);   // reuse k as q index
            }
            __syncthreads();

            // ---- 16x 16x16 WMMA tiles, operands via ds_load_b128 ----
            for (int t = wave; t < 16; t += 8) {
                const int qi = t >> 2, ki = t & 3;
                const __bf16* arow = &s_qfb[qi * 16 + lrow][0];
                const __bf16* brow = &s_kf [ki * 16 + lrow][0];
                v8f acc;
                for (int i = 0; i < 8; ++i) acc[i] = 0.f;
                for (int ks = 0; ks < 2; ++ks) {
                    const v16bf av = ld_a_op(arow, ks, half);
                    const v16bf bv = ld_b_op(brow, ks, half);
                    acc = __builtin_amdgcn_wmma_f32_16x16x32_bf16(
                        false, av, false, bv, (short)0, acc, false, false);
                }
                for (int r = 0; r < 8; ++r) {
                    const int qr = m * QQ + qi * 16 + r + half * 8;
                    const int kc = n * QQ + ki * 16 + lrow;
                    T[obase + (size_t)qr * LL + kc] = acc[r];
                }
            }
        } else {
            // ---- diagonal chunk (n == m) ----
            // strictly-lower 16x16 sub-tiles via WMMA (waves 0..5)
            const int qis[6] = {1, 2, 2, 3, 3, 3};
            const int kis[6] = {0, 0, 1, 0, 1, 2};
            if (wave < 6) {
                const int qi = qis[wave], ki = kis[wave];
                const int q      = qi * 16 + lrow;       // within chunk, q >= 16
                const int kcol   = ki * 16 + lrow;
                const int refrow = ki * 16 + 15;         // sub-chunk boundary ref
                v8f acc;
                for (int i = 0; i < 8; ++i) acc[i] = 0.f;
                for (int ks = 0; ks < 2; ++ks) {
                    v16bf av, bv;
                    for (int e = 0; e < 16; ++e) {
                        const int da = ks * 32 + a_kmap(e, half);
                        const float cg  = C[base + (size_t)(m * QQ + q) * DH + da];
                        const float sm1 = s_s[q - 1][da];
                        av[e] = f2bf(cg * __expf(sm1 - s_s[refrow][da]));     // exp(<=0)
                        const int db = ks * 32 + b_kmap(e, half);
                        const float bg = Bf[base + (size_t)(m * QQ + kcol) * DH + db];
                        bv[e] = f2bf(bg * __expf(s_s[refrow][db] - s_s[kcol][db]));
                    }
                    acc = __builtin_amdgcn_wmma_f32_16x16x32_bf16(
                        false, av, false, bv, (short)0, acc, false, false);
                }
                for (int r = 0; r < 8; ++r) {
                    const int qr = m * QQ + qi * 16 + r + half * 8;
                    const int kc = m * QQ + ki * 16 + lrow;
                    T[obase + (size_t)qr * LL + kc] = acc[r];
                }
            }
            // diagonal sub-tiles (masked element-wise) + upper sub-tiles (zero)
            for (int st = 0; st < 16; ++st) {
                const int qi = st >> 2, ki = st & 3;
                if (qi > ki) continue;                   // WMMA handled above
                const int ql = qi * 16 + (tid >> 4);
                const int kl = ki * 16 + (tid & 15);
                float val = 0.f;
                if (qi == ki) {
                    if (ql == kl) {
                        float s = 0.f;
                        for (int d = 0; d < DH; ++d)
                            s += C[base + (size_t)(m * QQ + ql) * DH + d] *
                                 Bf[base + (size_t)(m * QQ + ql) * DH + d];
                        val = s;
                    } else if (ql > kl) {
                        float s = 0.f;
                        for (int d = 0; d < DH; ++d) {
                            const float cg  = C[base + (size_t)(m * QQ + ql) * DH + d];
                            const float bg  = Bf[base + (size_t)(m * QQ + kl) * DH + d];
                            const float sm1 = s_s[ql - 1][d];   // ql >= 1 here
                            s += cg * bg * __expf(sm1 - s_s[kl][d]);
                        }
                        val = s;
                    }
                }
                T[obase + (size_t)(m * QQ + ql) * LL + (m * QQ + kl)] = val;
            }
        }
    }

    // zero the upper-triangular chunk tiles (n > m) with 128-bit stores
    for (int n = m + 1; n < NC; ++n) {
        for (int idx = tid; idx < QQ * QQ / 4; idx += 256) {
            const int q  = idx >> 4;
            const int k4 = (idx & 15) * 4;
            const float4 z = make_float4(0.f, 0.f, 0.f, 0.f);
            *(float4*)&T[obase + (size_t)(m * QQ + q) * LL + (n * QQ + k4)] = z;
        }
    }
}

// ---------------------------------------------------------------------------
// Host launcher
// ---------------------------------------------------------------------------
extern "C" void kernel_launch(void* const* d_in, const int* in_sizes, int n_in,
                              void* d_out, int out_size, void* d_ws, size_t ws_size,
                              hipStream_t stream) {
    const float* x   = (const float*)d_in[0];
    const float* W_A = (const float*)d_in[1];
    const float* b_A = (const float*)d_in[2];
    const float* W_B = (const float*)d_in[3];
    const float* b_B = (const float*)d_in[4];
    const float* W_C = (const float*)d_in[5];
    const float* b_C = (const float*)d_in[6];
    float* out = (float*)d_out;

    const size_t CH = (size_t)BB * HH * LL * DH;  // 1,048,576 floats per tensor
    float* A  = (float*)d_ws;                     // sigmoid(x W_A^T + b_A)
    float* Bf = A  + CH;
    float* C  = Bf + CH;
    float* S  = C  + CH;                          // cumsum(log A)

    // 3 projections: (2048 x 512) output in 64x64 tiles -> 32*8 = 256 tiles
    proj_kernel<<<dim3(256, 3), 256, 0, stream>>>(x, W_A, b_A, W_B, b_B, W_C, b_C,
                                                  A, Bf, C);
    scan_kernel<<<dim3(BB * HH), DH, 0, stream>>>(A, S);
    mix_kernel<<<dim3(NC, HH, BB), 256, 0, stream>>>(C, Bf, S, out);
}
